// EncoderLayer_57543971832496
// MI455X (gfx1250) — matrix operand order repaired and checked by reading, not compiled, not run on hardware
//
#include <hip/hip_runtime.h>

// ---------------- problem constants ----------------
#define B_   4
#define S_   2048
#define DM   1024
#define NH   16
#define DH   64
#define DFF  4096
#define MS   (B_ * S_)          // 8192 rows
#define EPS  1e-5f

// ---------------- WMMA types ----------------
typedef __attribute__((ext_vector_type(16))) __bf16 v16bf;
typedef __attribute__((ext_vector_type(8)))  float  v8f;
typedef __attribute__((ext_vector_type(4)))  unsigned v4u;
typedef __attribute__((ext_vector_type(8)))  unsigned v8u;

union Frag16 {
  v16bf v;
  unsigned u[8];
};

__device__ __forceinline__ unsigned short f2bf(float f) {
  unsigned u = __float_as_uint(f);
  return (unsigned short)((u + 0x7FFFu + ((u >> 16) & 1u)) >> 16);
}

// ---- CDNA5 async global->LDS copy (ASYNCcnt path) ----
__device__ __forceinline__ void async_copy_b128(unsigned lds_addr, const void* gaddr) {
  asm volatile("global_load_async_to_lds_b128 %0, %1, off"
               :: "v"(lds_addr), "v"(gaddr)
               : "memory");
}
__device__ __forceinline__ void wait_async() {
  asm volatile("s_wait_asynccnt 0x0" ::: "memory");
}
__device__ __forceinline__ unsigned lds_off(const void* p) {
  // generic pointer to LDS: low 32 bits are the LDS byte offset
  return (unsigned)(unsigned long long)p;
}

// ---- CDNA5 Tensor Data Mover: 2D tile (bf16) global -> LDS ----
// D# per CDNA5 ISA: group0 = {flags, lds_addr, global_addr, type},
// group1 = {data_size, tensor dims, tile dims, strides}; groups 2/3 unused (2D).
__device__ __forceinline__ void tdm_load_2d_bf16(
    unsigned lds_addr, const void* gaddr,
    unsigned tile_d0, unsigned tile_d1,          // tile: elems x rows
    unsigned tensor_d0, unsigned tensor_d1,      // tensor bounds (elems, rows)
    unsigned stride0)                            // row stride in elements
{
  unsigned long long ga = (unsigned long long)gaddr;
  v4u g0;
  g0[0] = 1u;                                              // count=1, user mode
  g0[1] = lds_addr;                                        // LDS byte address
  g0[2] = (unsigned)ga;                                    // global_addr[31:0]
  g0[3] = ((unsigned)(ga >> 32) & 0x01FFFFFFu) | 0x80000000u; // addr[56:32] | type=2
  v8u g1;
  g1[0] = (1u << 16);                                      // wg_mask=0, data_size=1 (2B)
  g1[1] = (tensor_d0 & 0xFFFFu) << 16;                     // tensor_dim0[15:0]
  g1[2] = (tensor_d0 >> 16) | ((tensor_d1 & 0xFFFFu) << 16); // dim0 hi | dim1 lo
  g1[3] = (tensor_d1 >> 16) | (tile_d0 << 16);             // dim1 hi | tile_dim0
  g1[4] = tile_d1;                                         // tile_dim1 | tile_dim2=0
  g1[5] = stride0;                                         // tensor_dim0_stride lo
  g1[6] = 0u;                                              // stride0 hi | dim1_stride lo
  g1[7] = 0u;
  v4u gz = {0u, 0u, 0u, 0u};
  asm volatile("tensor_load_to_lds %0, %1, %2, %3"
               :: "s"(g0), "s"(g1), "s"(gz), "s"(gz)
               : "memory");
}

// ---------------- fp32 -> bf16 convert ----------------
__global__ __launch_bounds__(256) void cvt_kernel(const float* __restrict__ in,
                                                  unsigned short* __restrict__ out,
                                                  int n) {
  int i = blockIdx.x * 256 + threadIdx.x;
  if (i < n) out[i] = f2bf(in[i]);
}

// ---------------- tiled GEMM: C[M,N] = act(A[M,K] @ W[K,N] + bias) ----------------
// 128 threads = 4 waves; block tile 64x64; wave tile 16(M) x 64(N); K-step 32.
// A tile staged row-major via async-to-LDS; W tile staged *transposed* so B
// fragments are two contiguous 16B runs per lane (2x ds_load_b128).
template <bool RELU>
__global__ __launch_bounds__(128) void gemm_bf16(
    const unsigned short* __restrict__ A,   // bf16 [M,K]
    const unsigned short* __restrict__ W,   // bf16 [K,N]
    const float* __restrict__ bias,         // [N]
    float* __restrict__ outF,               // optional fp32 [M,N]
    unsigned short* __restrict__ outH,      // optional bf16 [M,N]
    int M, int N, int K) {
  __shared__ __align__(16) unsigned short As[64 * 32];   // [m][k]
  __shared__ __align__(16) unsigned short BsT[64 * 32];  // [n][k] (transposed)

  const int t    = threadIdx.x;
  const int lane = t & 31;
  const int wave = t >> 5;
  const int ln   = lane & 15;
  const int half = lane >> 4;
  const int m0   = blockIdx.y * 64;
  const int n0   = blockIdx.x * 64;

  v8f acc[4];
  const v8f zero8 = {0.f, 0.f, 0.f, 0.f, 0.f, 0.f, 0.f, 0.f};
#pragma unroll
  for (int c = 0; c < 4; ++c) acc[c] = zero8;

  // per-thread staging geometry
  const int arow0 = t >> 2;            // A pass0: rows 0..31
  const int acol  = (t & 3) * 8;       // 8 bf16 = 16B
  const int wk    = t >> 2;            // W row (k) 0..31
  const int wn0   = (t & 3) * 16;      // 16 columns handled per thread

  for (int k0 = 0; k0 < K; k0 += 32) {
    // ---- A tile (64x32): async global->LDS, 16B per lane, 2 passes ----
#pragma unroll
    for (int p = 0; p < 2; ++p) {
      int row = arow0 + p * 32;
      async_copy_b128(lds_off(&As[row * 32 + acol]),
                      A + (size_t)(m0 + row) * K + k0 + acol);
    }
    // ---- W tile (32x64) staged transposed into BsT[n][k] ----
    {
      const unsigned short* wsrc = W + (size_t)(k0 + wk) * N + n0 + wn0;
      float4 w0 = *reinterpret_cast<const float4*>(wsrc);
      float4 w1 = *reinterpret_cast<const float4*>(wsrc + 8);
      const unsigned short* e0 = reinterpret_cast<const unsigned short*>(&w0);
      const unsigned short* e1 = reinterpret_cast<const unsigned short*>(&w1);
#pragma unroll
      for (int e = 0; e < 8; ++e) {
        BsT[(wn0 + e) * 32 + wk]     = e0[e];
        BsT[(wn0 + 8 + e) * 32 + wk] = e1[e];
      }
      if (k0 + 32 < K)
        __builtin_prefetch(W + (size_t)(k0 + 32 + wk) * N + n0 + wn0, 0, 1);
    }
    wait_async();
    __syncthreads();

    // ---- A fragment (ISA 16-bit A layout: lanes0-15 K in {0..7,16..23}, lanes16-31 +8) ----
    Frag16 a;
    {
      int m  = wave * 16 + ln;
      int kb = half ? 8 : 0;
#pragma unroll
      for (int j = 0; j < 4; ++j)
        a.u[j] = *reinterpret_cast<const unsigned*>(&As[m * 32 + kb + 2 * j]);
#pragma unroll
      for (int j = 0; j < 4; ++j)
        a.u[4 + j] = *reinterpret_cast<const unsigned*>(&As[m * 32 + kb + 16 + 2 * j]);
    }

    // ---- 4 B fragments + WMMAs (B layout: lanes0-15 K in {0..7,16..23}, lanes16-31 +8) ----
#pragma unroll
    for (int c = 0; c < 4; ++c) {
      Frag16 bf;
      int n  = c * 16 + ln;
      int kb = half ? 8 : 0;
#pragma unroll
      for (int j = 0; j < 8; ++j)
        bf.u[j] = *reinterpret_cast<const unsigned*>(&BsT[n * 32 + kb + (j & 3) * 2 + (j >> 2) * 16]);
      acc[c] = __builtin_amdgcn_wmma_f32_16x16x32_bf16(
          false, a.v, false, bf.v, (short)0, acc[c], false, false);
    }
    __syncthreads();
  }

  // ---- epilogue: bias, activation, store ----
#pragma unroll
  for (int c = 0; c < 4; ++c) {
    int col  = n0 + c * 16 + ln;
    float bv = bias ? bias[col] : 0.f;
#pragma unroll
    for (int j = 0; j < 8; ++j) {
      int row = m0 + wave * 16 + j + 8 * half;
      float v = acc[c][j] + bv;
      if (RELU) v = fmaxf(v, 0.f);
      size_t off = (size_t)row * N + col;
      if (outF) outF[off] = v;
      if (outH) outH[off] = f2bf(v);
    }
  }
}

// ---------------- flash attention: one wave per (b, h, 16-row Q tile) ----------------
// K/V tiles staged with the Tensor Data Mover (TENSORcnt path).
__global__ __launch_bounds__(32) void attn_kernel(
    const unsigned short* __restrict__ Q,   // bf16 [B*S, DM]
    const unsigned short* __restrict__ Km,  // bf16 [B*S, DM]
    const unsigned short* __restrict__ Vm,  // bf16 [B*S, DM]
    const int* __restrict__ mask,           // [B, S]
    unsigned short* __restrict__ O) {       // bf16 [B*S, DM]
  __shared__ __align__(16) unsigned short Ks[32 * 64];
  __shared__ __align__(16) unsigned short Vs[32 * 64];
  __shared__ __align__(16) unsigned short Ps[16 * 32];

  const int lane = threadIdx.x;
  const int ln   = lane & 15;
  const int half = lane >> 4;
  const int q0   = blockIdx.x * 16;
  const int h    = blockIdx.y;
  const int b    = blockIdx.z;

  // Q tile -> two A fragments (dh 0..31, 32..63)
  Frag16 qa[2];
  {
    const unsigned short* qrow = Q + (size_t)(b * S_ + q0 + ln) * DM + h * DH;
    int kb = half ? 8 : 0;
#pragma unroll
    for (int c = 0; c < 2; ++c) {
#pragma unroll
      for (int j = 0; j < 4; ++j)
        qa[c].u[j] = *reinterpret_cast<const unsigned*>(qrow + c * 32 + kb + 2 * j);
#pragma unroll
      for (int j = 0; j < 4; ++j)
        qa[c].u[4 + j] =
            *reinterpret_cast<const unsigned*>(qrow + c * 32 + kb + 16 + 2 * j);
    }
  }

  const v8f zero8 = {0.f, 0.f, 0.f, 0.f, 0.f, 0.f, 0.f, 0.f};
  v8f oacc[4];
#pragma unroll
  for (int c = 0; c < 4; ++c) oacc[c] = zero8;
  float rmax[8], rsum[8];
#pragma unroll
  for (int j = 0; j < 8; ++j) { rmax[j] = -3.4e38f; rsum[j] = 0.f; }

  const float scale = 0.125f;  // 1/sqrt(64)

  for (int kv0 = 0; kv0 < S_; kv0 += 32) {
    // ---- stage 32x64 K and V blocks via Tensor Data Mover ----
    // tile: 64 elems x 32 rows, row stride DM elems; LDS dest packed row-major.
    tdm_load_2d_bf16(lds_off(Ks),
                     Km + (size_t)(b * S_ + kv0) * DM + h * DH,
                     64u, 32u, 64u, (unsigned)S_, (unsigned)DM);
    tdm_load_2d_bf16(lds_off(Vs),
                     Vm + (size_t)(b * S_ + kv0) * DM + h * DH,
                     64u, 32u, 64u, (unsigned)S_, (unsigned)DM);
    __builtin_amdgcn_s_wait_tensorcnt(0);
    __syncthreads();

    // ---- scores: two 16x16 tiles, each accumulated over dh in two k=32 steps ----
    v8f s0 = zero8, s1 = zero8;
#pragma unroll
    for (int tile = 0; tile < 2; ++tile) {
      v8f c = zero8;
#pragma unroll
      for (int ch = 0; ch < 2; ++ch) {
        Frag16 bf;
        int kb = half ? 8 : 0;
#pragma unroll
        for (int j = 0; j < 8; ++j)
          bf.u[j] = *reinterpret_cast<const unsigned*>(
              &Ks[(tile * 16 + ln) * 64 + ch * 32 + kb + (j & 3) * 2 + (j >> 2) * 16]);
        c = __builtin_amdgcn_wmma_f32_16x16x32_bf16(
            false, qa[ch].v, false, bf.v, (short)0, c, false, false);
      }
      if (tile == 0) s0 = c; else s1 = c;
    }

    // ---- scale + mask ----
    bool mk0 = mask[(size_t)b * S_ + kv0 + ln] != 0;
    bool mk1 = mask[(size_t)b * S_ + kv0 + 16 + ln] != 0;
#pragma unroll
    for (int j = 0; j < 8; ++j) {
      s0[j] = mk0 ? s0[j] * scale : -1e9f;
      s1[j] = mk1 ? s1[j] * scale : -1e9f;
    }

    // ---- online softmax (row = VGPR j, spread over 16 lanes of this half) ----
    float alpha[8];
#pragma unroll
    for (int j = 0; j < 8; ++j) {
      float v = fmaxf(s0[j], s1[j]);
#pragma unroll
      for (int d = 1; d < 16; d <<= 1) v = fmaxf(v, __shfl_xor(v, d, 32));
      float nm = fmaxf(rmax[j], v);
      alpha[j] = __expf(rmax[j] - nm);
      rmax[j]  = nm;
      float p0 = __expf(s0[j] - nm);
      float p1 = __expf(s1[j] - nm);
      s0[j] = p0; s1[j] = p1;
      float rs = p0 + p1;
#pragma unroll
      for (int d = 1; d < 16; d <<= 1) rs += __shfl_xor(rs, d, 32);
      rsum[j] = rsum[j] * alpha[j] + rs;
    }
#pragma unroll
    for (int c = 0; c < 4; ++c)
#pragma unroll
      for (int j = 0; j < 8; ++j) oacc[c][j] = oacc[c][j] * alpha[j];

    // ---- P (C-layout) -> LDS -> A-layout fragment ----
#pragma unroll
    for (int j = 0; j < 8; ++j) {
      int row = j + 8 * half;
      Ps[row * 32 + ln]      = f2bf(s0[j]);
      Ps[row * 32 + 16 + ln] = f2bf(s1[j]);
    }
    __syncthreads();

    Frag16 pa;
    {
      int kb = half ? 8 : 0;
#pragma unroll
      for (int j = 0; j < 4; ++j)
        pa.u[j] = *reinterpret_cast<const unsigned*>(&Ps[ln * 32 + kb + 2 * j]);
#pragma unroll
      for (int j = 0; j < 4; ++j)
        pa.u[4 + j] =
            *reinterpret_cast<const unsigned*>(&Ps[ln * 32 + kb + 16 + 2 * j]);
    }

    // ---- O += P @ V (4 n-chunks of 16 dh columns) ----
#pragma unroll
    for (int c = 0; c < 4; ++c) {
      Frag16 vb;
      int kb = half ? 8 : 0;
#pragma unroll
      for (int j = 0; j < 8; ++j) {
        unsigned lo = Vs[(kb + (j & 3) * 2 + (j >> 2) * 16) * 64 + c * 16 + ln];
        unsigned hi = Vs[(kb + (j & 3) * 2 + (j >> 2) * 16 + 1) * 64 + c * 16 + ln];
        vb.u[j] = lo | (hi << 16);
      }
      oacc[c] = __builtin_amdgcn_wmma_f32_16x16x32_bf16(
          false, pa.v, false, vb.v, (short)0, oacc[c], false, false);
    }
    __syncthreads();
  }

  // ---- normalize and store (bf16, [B*S, DM] layout for the O-projection GEMM) ----
#pragma unroll
  for (int c = 0; c < 4; ++c)
#pragma unroll
    for (int j = 0; j < 8; ++j) {
      int row = q0 + j + 8 * half;
      O[(size_t)(b * S_ + row) * DM + h * DH + c * 16 + ln] =
          f2bf(oacc[c][j] / rsum[j]);
    }
}

// ---------------- residual + layernorm ----------------
__global__ __launch_bounds__(256) void ln_kernel(
    const float* __restrict__ a, const float* __restrict__ r,
    const float* __restrict__ g, const float* __restrict__ be,
    float* __restrict__ outF, unsigned short* __restrict__ outH) {
  __shared__ float red[16];
  const int t = threadIdx.x;
  const size_t row = blockIdx.x;
  const float* pa = a + row * DM;
  const float* pr = r + row * DM;

  float v[4];
  float s = 0.f, sq = 0.f;
#pragma unroll
  for (int i = 0; i < 4; ++i) {
    int col = i * 256 + t;
    float x = pa[col] + pr[col];
    v[i] = x; s += x; sq += x * x;
  }
#pragma unroll
  for (int d = 1; d < 32; d <<= 1) {
    s  += __shfl_xor(s, d, 32);
    sq += __shfl_xor(sq, d, 32);
  }
  if ((t & 31) == 0) { red[t >> 5] = s; red[8 + (t >> 5)] = sq; }
  __syncthreads();
  float ts = 0.f, tsq = 0.f;
#pragma unroll
  for (int w = 0; w < 8; ++w) { ts += red[w]; tsq += red[8 + w]; }
  float mu   = ts * (1.f / DM);
  float var  = tsq * (1.f / DM) - mu * mu;
  float rstd = rsqrtf(var + EPS);
#pragma unroll
  for (int i = 0; i < 4; ++i) {
    int col = i * 256 + t;
    float y = (v[i] - mu) * rstd * g[col] + be[col];
    outF[row * DM + col] = y;
    if (outH) outH[row * DM + col] = f2bf(y);
  }
}

// ---------------- host-side orchestration ----------------
extern "C" void kernel_launch(void* const* d_in, const int* in_sizes, int n_in,
                              void* d_out, int out_size, void* d_ws, size_t ws_size,
                              hipStream_t stream) {
  (void)in_sizes; (void)n_in; (void)out_size; (void)ws_size;

  const float* x  = (const float*)d_in[0];
  const float* Wq = (const float*)d_in[1];  const float* bq = (const float*)d_in[2];
  const float* Wk = (const float*)d_in[3];  const float* bk = (const float*)d_in[4];
  const float* Wv = (const float*)d_in[5];  const float* bv = (const float*)d_in[6];
  const float* Wo = (const float*)d_in[7];  const float* bo = (const float*)d_in[8];
  const float* W1 = (const float*)d_in[9];  const float* b1 = (const float*)d_in[10];
  const float* W2 = (const float*)d_in[11]; const float* b2 = (const float*)d_in[12];
  const float* g1 = (const float*)d_in[13]; const float* be1 = (const float*)d_in[14];
  const float* g2 = (const float*)d_in[15]; const float* be2 = (const float*)d_in[16];
  const int* mask = (const int*)d_in[17];
  float* out = (float*)d_out;

  // bump allocator over d_ws
  char* ws = (char*)d_ws;
  size_t off = 0;
  auto alloc = [&](size_t bytes) -> void* {
    void* p = ws + off;
    off = (off + bytes + 255) & ~(size_t)255;
    return p;
  };

  unsigned short* xb  = (unsigned short*)alloc((size_t)MS * DM * 2);
  unsigned short* Wqb = (unsigned short*)alloc((size_t)DM * DM * 2);
  unsigned short* Wkb = (unsigned short*)alloc((size_t)DM * DM * 2);
  unsigned short* Wvb = (unsigned short*)alloc((size_t)DM * DM * 2);
  unsigned short* Wob = (unsigned short*)alloc((size_t)DM * DM * 2);
  unsigned short* W1b = (unsigned short*)alloc((size_t)DM * DFF * 2);
  unsigned short* W2b = (unsigned short*)alloc((size_t)DFF * DM * 2);
  unsigned short* Qb  = (unsigned short*)alloc((size_t)MS * DM * 2);
  unsigned short* Kb  = (unsigned short*)alloc((size_t)MS * DM * 2);
  unsigned short* Vb  = (unsigned short*)alloc((size_t)MS * DM * 2);
  unsigned short* Ob  = (unsigned short*)alloc((size_t)MS * DM * 2);
  float*          attnf = (float*)alloc((size_t)MS * DM * 4);
  float*          h1f   = (float*)alloc((size_t)MS * DM * 4);
  unsigned short* h1b   = (unsigned short*)alloc((size_t)MS * DM * 2);
  unsigned short* f1b   = (unsigned short*)alloc((size_t)MS * DFF * 2);
  float*          f2f   = (float*)alloc((size_t)MS * DM * 4);

  auto cvt = [&](const float* src, unsigned short* dst, int n) {
    cvt_kernel<<<(n + 255) / 256, 256, 0, stream>>>(src, dst, n);
  };
  cvt(x,  xb,  MS * DM);
  cvt(Wq, Wqb, DM * DM);
  cvt(Wk, Wkb, DM * DM);
  cvt(Wv, Wvb, DM * DM);
  cvt(Wo, Wob, DM * DM);
  cvt(W1, W1b, DM * DFF);
  cvt(W2, W2b, DFF * DM);

  // QKV projections (bf16 out for attention)
  dim3 gQ(DM / 64, MS / 64);
  gemm_bf16<false><<<gQ, 128, 0, stream>>>(xb, Wqb, bq, nullptr, Qb, MS, DM, DM);
  gemm_bf16<false><<<gQ, 128, 0, stream>>>(xb, Wkb, bk, nullptr, Kb, MS, DM, DM);
  gemm_bf16<false><<<gQ, 128, 0, stream>>>(xb, Wvb, bv, nullptr, Vb, MS, DM, DM);

  // attention
  dim3 gA(S_ / 16, NH, B_);
  attn_kernel<<<gA, 32, 0, stream>>>(Qb, Kb, Vb, mask, Ob);

  // output projection (fp32 out)
  gemm_bf16<false><<<gQ, 128, 0, stream>>>(Ob, Wob, bo, attnf, nullptr, MS, DM, DM);

  // LN1: x + attn_out
  ln_kernel<<<MS, 256, 0, stream>>>(x, attnf, g1, be1, h1f, h1b);

  // FFN
  dim3 gF1(DFF / 64, MS / 64);
  gemm_bf16<true><<<gF1, 128, 0, stream>>>(h1b, W1b, b1, nullptr, f1b, MS, DFF, DM);
  gemm_bf16<false><<<gQ, 128, 0, stream>>>(f1b, W2b, b2, f2f, nullptr, MS, DM, DFF);

  // LN2: h1 + ff -> output
  ln_kernel<<<MS, 256, 0, stream>>>(h1f, f2f, g2, be2, out, nullptr);
}
